// GPTMultiheadMaskedAttention_18562848654002
// MI455X (gfx1250) — compile-verified, matrix operand order checked by reading, not run on hardware
//
#include <hip/hip_runtime.h>
#include <hip/hip_bf16.h>
#include <math.h>

// ---------------------------------------------------------------------------
// CDNA5 WMMA (wave32): bf16 A/B fragments (16 elems/lane = 8 VGPRs),
// f32 C/D fragments (8 elems/lane = 8 VGPRs).
// ---------------------------------------------------------------------------
typedef __attribute__((ext_vector_type(16))) __bf16 v16bf;
typedef __attribute__((ext_vector_type(8)))  __bf16 v8bf;
typedef __attribute__((ext_vector_type(8)))  float  v8f;
typedef __attribute__((__vector_size__(4 * sizeof(int)))) int v4i;

static constexpr int S_LEN  = 2048;
static constexpr int HID    = 2048;
static constexpr int NH     = 16;
static constexpr int HD     = 128;
static constexpr int N_QKV  = 3 * HID;           // 6144
static constexpr int M_ROWS = 2 * S_LEN;         // 4096 (B*S)

__device__ __forceinline__ v8f wmma_bf16(v16bf a, v16bf b, v8f c) {
    // v_wmma_f32_16x16x32_bf16: (neg_a, A, neg_b, B, c_mod, C, reuse_a, reuse_b)
    return __builtin_amdgcn_wmma_f32_16x16x32_bf16(false, a, false, b,
                                                   (short)0, c, false, false);
}

// CDNA5 async global->LDS copy (GLOBAL_LOAD_ASYNC_TO_LDS_B128, ASYNCcnt).
// No VGPR transit: data lands in LDS directly.
// Signature per hipcc diagnostic: param 1 is v4i in addrspace(1) ("__device__").
__device__ __forceinline__ void async_b128(const void* g, void* l) {
    __builtin_amdgcn_global_load_async_to_lds_b128(
        (__attribute__((address_space(1))) v4i*)(void*)g,
        (__attribute__((address_space(3))) v4i*)l,
        0, 0);
}
#define WAIT_ASYNC_0() asm volatile("s_wait_asynccnt 0" ::: "memory")
#define WAIT_ASYNC_8() asm volatile("s_wait_asynccnt 8" ::: "memory")

// 16x32 bf16 A/B fragment, ISA lane layout: lanes 0-15 take K chunks
// [0..7],[16..23]; lanes 16-31 take [8..15],[24..31].
// Caller passes p = &tile[row(lane) * ld + (lane>>4)*8].
__device__ __forceinline__ v16bf load_frag16(const __bf16* p) {
    v8bf lo = *(const v8bf*)(p);
    v8bf hi = *(const v8bf*)(p + 16);
    v16bf r;
#pragma unroll
    for (int i = 0; i < 8; ++i) { r[i] = lo[i]; r[i + 8] = hi[i]; }
    return r;
}

__device__ __forceinline__ v8f zero_v8f() {
    v8f z;
#pragma unroll
    for (int i = 0; i < 8; ++i) z[i] = 0.0f;
    return z;
}

// ---------------------------------------------------------------------------
// Kernel 1a: f32 -> bf16 streaming convert (for activations x)
// ---------------------------------------------------------------------------
__global__ void cvt_f32_bf16(const float* __restrict__ src,
                             __bf16* __restrict__ dst, int n) {
    int i = blockIdx.x * blockDim.x + threadIdx.x;
    int stride = gridDim.x * blockDim.x;
    for (; i < n; i += stride) dst[i] = (__bf16)src[i];
}

// ---------------------------------------------------------------------------
// Kernel 1b: f32 [K][N] -> bf16 transposed [N][K] (one-shot weight prep so
// the GEMM hot loop needs no LDS scatter). 64x64 tiles via LDS.
// ---------------------------------------------------------------------------
__global__ __launch_bounds__(256)
void cvt_transpose_bf16(const float* __restrict__ src, __bf16* __restrict__ dst,
                        int K, int N) {
    __shared__ __bf16 t[64][65];
    const int k0 = blockIdx.y * 64, n0 = blockIdx.x * 64;
    const int tid = threadIdx.x;
#pragma unroll
    for (int i = 0; i < 16; ++i) {
        int c = tid + 256 * i;
        int r = c >> 6, col = c & 63;
        t[r][col] = (__bf16)src[(size_t)(k0 + r) * N + n0 + col];
    }
    __syncthreads();
#pragma unroll
    for (int i = 0; i < 16; ++i) {
        int c = tid + 256 * i;
        int r = c >> 6, col = c & 63;
        dst[(size_t)(n0 + r) * K + k0 + col] = t[col][r];
    }
}

// ---------------------------------------------------------------------------
// Kernel 2/4: 128x128 tiled bf16 GEMM, B pre-transposed [N][K].
// 128 threads = 4 waves, each computing a 64x64 sub-tile (4x4 fragments):
// 16 v_wmma per wave per K-step. Double-buffered LDS fed by async
// global->LDS b128 copies (ASYNCcnt), so no staging VGPRs are held live.
// MODE 0: f32 output + bias. MODE 1: QKV scatter (V stored transposed).
// ---------------------------------------------------------------------------
template <int MODE>
__global__ __launch_bounds__(128)
void gemm_bf16_wmma(const __bf16* __restrict__ A, const __bf16* __restrict__ Bt,
                    const float* __restrict__ bias, int M, int N, int K,
                    float* __restrict__ outF,
                    __bf16* __restrict__ outQ, __bf16* __restrict__ outK,
                    __bf16* __restrict__ outV) {
    constexpr int LDT = 48;                         // 96 B rows (16B-aligned)
    __shared__ alignas(16) __bf16 AsBuf[2][128 * LDT];
    __shared__ alignas(16) __bf16 BsBuf[2][128 * LDT];

    const int tid = threadIdx.x;
    const int m0 = blockIdx.y * 128, n0 = blockIdx.x * 128;
    const int w = tid >> 5, l = tid & 31;
    const int ln = l & 15, lh = l >> 4;
    const int wm = w >> 1, wn = w & 1;              // wave tile origin (64x64)

    // Staging map: 4 x 16B chunks per thread per tile (128 rows x 32 cols).
    int rs[4], ks8[4];
#pragma unroll
    for (int i = 0; i < 4; ++i) {
        int c = tid + 128 * i;
        rs[i] = c >> 2;
        ks8[i] = (c & 3) * 8;
    }
    // 8 async b128 copies per thread per tile: global -> LDS directly.
    auto issue = [&](int buf, int k0) {
#pragma unroll
        for (int i = 0; i < 4; ++i) {
            async_b128(&A[(size_t)(m0 + rs[i]) * K + k0 + ks8[i]],
                       &AsBuf[buf][rs[i] * LDT + ks8[i]]);
            async_b128(&Bt[(size_t)(n0 + rs[i]) * K + k0 + ks8[i]],
                       &BsBuf[buf][rs[i] * LDT + ks8[i]]);
        }
    };

    v8f acc[4][4];
#pragma unroll
    for (int i = 0; i < 4; ++i)
#pragma unroll
        for (int j = 0; j < 4; ++j) acc[i][j] = zero_v8f();

    issue(0, 0);
    const int KSTEPS = K / 32;
    for (int st = 0; st < KSTEPS; ++st) {
        const int cur = st & 1;
        const bool more = (st + 1 < KSTEPS);
        if (more) {
            issue(cur ^ 1, (st + 1) * 32);   // prefetch next tile into other buffer
            WAIT_ASYNC_8();                  // our batch for `cur` has landed
        } else {
            WAIT_ASYNC_0();
        }
        __syncthreads();                     // all waves' `cur` data visible

        const __bf16* As = AsBuf[cur];
        const __bf16* Bs = BsBuf[cur];
        v16bf af[4];
#pragma unroll
        for (int fm = 0; fm < 4; ++fm)
            af[fm] = load_frag16(&As[(wm * 64 + fm * 16 + ln) * LDT + lh * 8]);
#pragma unroll
        for (int fn = 0; fn < 4; ++fn) {
            v16bf bf = load_frag16(&Bs[(wn * 64 + fn * 16 + ln) * LDT + lh * 8]);
#pragma unroll
            for (int fm = 0; fm < 4; ++fm)
                acc[fm][fn] = wmma_bf16(af[fm], bf, acc[fm][fn]);
        }
        __syncthreads();                     // done reading before refill
    }

    // Epilogue. C layout: reg r, lanes 0-15 -> (M=r, N=lane); lanes 16-31 -> M=r+8.
#pragma unroll
    for (int fm = 0; fm < 4; ++fm)
#pragma unroll
        for (int fn = 0; fn < 4; ++fn)
#pragma unroll
            for (int r = 0; r < 8; ++r) {
                int row = m0 + wm * 64 + fm * 16 + r + lh * 8;
                int col = n0 + wn * 64 + fn * 16 + ln;
                float v = acc[fm][fn][r] + bias[col];
                if constexpr (MODE == 0) {
                    outF[(size_t)row * N + col] = v;
                } else {
                    int which = col >> 11;          // 0:Q 1:K 2:V
                    int cc = col & 2047;
                    int h = cc >> 7, d = cc & 127;
                    int b = row >> 11, s = row & 2047;
                    if (which == 2) {               // V transposed: [b,h,d,s]
                        outV[(((size_t)(b * NH + h)) * HD + d) * S_LEN + s] = (__bf16)v;
                    } else {
                        __bf16* dst = (which == 0) ? outQ : outK;
                        dst[(((size_t)(b * NH + h)) * S_LEN + s) * HD + d] = (__bf16)v;
                    }
                }
            }
}

// ---------------------------------------------------------------------------
// Kernel 3: causal flash attention, one (b,h) x 64-query tile per block.
// 128 threads = 4 waves; each wave owns 16 query rows, Q in registers.
// K/V tiles staged with async global->LDS copies; per 64-key tile:
// QK^T (16 wmma), mask+scale, online softmax (shfl_xor within 16-lane
// halves), P relayout via per-wave LDS, O += P V (16 wmma).
// ---------------------------------------------------------------------------
__global__ __launch_bounds__(128)
void attn_flash_wmma(const __bf16* __restrict__ Qg, const __bf16* __restrict__ Kg,
                     const __bf16* __restrict__ Vtg, __bf16* __restrict__ attn) {
    constexpr int LKS = 136;   // 272 B rows
    constexpr int LVT = 72;    // 144 B rows
    constexpr int LP  = 72;
    __shared__ alignas(16) __bf16 Ks[64 * LKS];     // [key][d]   (Bt for Q.K^T)
    __shared__ alignas(16) __bf16 Vt[128 * LVT];    // [d][key]   (Bt for P.V)
    __shared__ alignas(16) __bf16 Pb[4 * 16 * LP];  // per-wave 16x64 probs

    const float SCALE = 0.08838834764831843f;       // 1/sqrt(128)
    const int qt = blockIdx.x, bh = blockIdx.y;
    const int q0 = qt * 64;
    const int tid = threadIdx.x, w = tid >> 5, l = tid & 31;
    const int ln = l & 15, lh = l >> 4;
    const size_t base = (size_t)bh * S_LEN * HD;    // same for [s][d] and [d][s]

    v16bf qf[4];
    {
        const __bf16* qp = Qg + base + (size_t)(q0 + w * 16 + ln) * HD + lh * 8;
#pragma unroll
        for (int ds = 0; ds < 4; ++ds) qf[ds] = load_frag16(qp + ds * 32);
    }

    v8f o[8];
#pragma unroll
    for (int i = 0; i < 8; ++i) o[i] = zero_v8f();
    float mrun[8], lrun[8];
#pragma unroll
    for (int r = 0; r < 8; ++r) { mrun[r] = -__builtin_inff(); lrun[r] = 0.0f; }

    for (int kt = 0; kt <= qt; ++kt) {
        const int k0 = kt * 64;
        // K tile [64][128] and V tile [128][64]: async b128 copies to LDS.
#pragma unroll
        for (int i = 0; i < 8; ++i) {
            int c = tid + 128 * i;
            int row = c >> 4, c8 = (c & 15) * 8;
            async_b128(&Kg[base + (size_t)(k0 + row) * HD + c8],
                       &Ks[row * LKS + c8]);
        }
#pragma unroll
        for (int i = 0; i < 8; ++i) {
            int c = tid + 128 * i;
            int d = c >> 3, k8 = (c & 7) * 8;
            async_b128(&Vtg[base + (size_t)d * S_LEN + k0 + k8],
                       &Vt[d * LVT + k8]);
        }
        WAIT_ASYNC_0();
        __syncthreads();

        // scores: 16 q-rows x 64 keys
        v8f s[4];
#pragma unroll
        for (int kc = 0; kc < 4; ++kc) {
            s[kc] = zero_v8f();
#pragma unroll
            for (int ds = 0; ds < 4; ++ds)
                s[kc] = wmma_bf16(qf[ds],
                    load_frag16(&Ks[(kc * 16 + ln) * LKS + ds * 32 + lh * 8]),
                    s[kc]);
        }
        // causal mask (diagonal tile only) + scale
        const bool diag = (kt == qt);
#pragma unroll
        for (int kc = 0; kc < 4; ++kc)
#pragma unroll
            for (int r = 0; r < 8; ++r) {
                float x = s[kc][r] * SCALE;
                if (diag && (k0 + kc * 16 + ln) > (q0 + w * 16 + r + lh * 8))
                    x = -__builtin_inff();
                s[kc][r] = x;
            }
        // online softmax stats (row lives across the 16 lanes of a half)
        float corr[8];
#pragma unroll
        for (int r = 0; r < 8; ++r) {
            float rm = s[0][r];
#pragma unroll
            for (int kc = 1; kc < 4; ++kc) rm = fmaxf(rm, s[kc][r]);
#pragma unroll
            for (int off = 8; off >= 1; off >>= 1)
                rm = fmaxf(rm, __shfl_xor(rm, off, 32));
            float mnew = fmaxf(mrun[r], rm);
            corr[r] = __expf(mrun[r] - mnew);
            float ps = 0.0f;
#pragma unroll
            for (int kc = 0; kc < 4; ++kc) {
                float p = __expf(s[kc][r] - mnew);
                s[kc][r] = p;
                ps += p;
            }
#pragma unroll
            for (int off = 8; off >= 1; off >>= 1)
                ps += __shfl_xor(ps, off, 32);
            lrun[r] = lrun[r] * corr[r] + ps;
            mrun[r] = mnew;
        }
#pragma unroll
        for (int dc = 0; dc < 8; ++dc)
#pragma unroll
            for (int r = 0; r < 8; ++r) o[dc][r] *= corr[r];

        // P: C-frag layout -> LDS -> A fragments.
        __bf16* Pw = &Pb[w * 16 * LP];
#pragma unroll
        for (int kc = 0; kc < 4; ++kc)
#pragma unroll
            for (int r = 0; r < 8; ++r)
                Pw[(r + lh * 8) * LP + kc * 16 + ln] = (__bf16)s[kc][r];
        __syncthreads();

        // O += P V
#pragma unroll
        for (int ks = 0; ks < 2; ++ks) {
            v16bf pf = load_frag16(&Pw[ln * LP + ks * 32 + lh * 8]);
#pragma unroll
            for (int dc = 0; dc < 8; ++dc)
                o[dc] = wmma_bf16(pf,
                    load_frag16(&Vt[(dc * 16 + ln) * LVT + ks * 32 + lh * 8]),
                    o[dc]);
        }
        __syncthreads();
    }

    // Normalize, emit attn [B][S][H] bf16 (heads concatenated).
    const int b = bh >> 4, h = bh & 15;
    float inv[8];
#pragma unroll
    for (int r = 0; r < 8; ++r) inv[r] = 1.0f / lrun[r];
#pragma unroll
    for (int dc = 0; dc < 8; ++dc)
#pragma unroll
        for (int r = 0; r < 8; ++r) {
            int qrow = q0 + w * 16 + r + lh * 8;
            attn[((size_t)(b * S_LEN + qrow)) * HID + h * HD + dc * 16 + ln] =
                (__bf16)(o[dc][r] * inv[r]);
        }
}

// ---------------------------------------------------------------------------
// Host launcher
// ---------------------------------------------------------------------------
extern "C" void kernel_launch(void* const* d_in, const int* in_sizes, int n_in,
                              void* d_out, int out_size, void* d_ws, size_t ws_size,
                              hipStream_t stream) {
    const float* x     = (const float*)d_in[0];   // [2,2048,2048]
    const float* W_qkv = (const float*)d_in[1];   // [2048,6144]
    const float* b_qkv = (const float*)d_in[2];   // [6144]
    const float* W_o   = (const float*)d_in[3];   // [2048,2048]
    const float* b_o   = (const float*)d_in[4];   // [2048]
    float* out = (float*)d_out;                   // [2,2048,2048]

    const size_t nX    = (size_t)M_ROWS * HID;
    const size_t nWqkv = (size_t)HID * N_QKV;
    const size_t nWo   = (size_t)HID * HID;
    const size_t nHead = (size_t)2 * NH * S_LEN * HD;

    char* p = (char*)d_ws;
    auto take = [&p](size_t elems) {
        __bf16* r = (__bf16*)p;
        p += (elems * sizeof(__bf16) + 255) & ~(size_t)255;
        return r;
    };
    __bf16* xb    = take(nX);
    __bf16* wqkvt = take(nWqkv);   // transposed [6144][2048]
    __bf16* wot   = take(nWo);     // transposed [2048][2048]
    __bf16* qb    = take(nHead);   // [b,h,s,d]
    __bf16* kb    = take(nHead);   // [b,h,s,d]
    __bf16* vbt   = take(nHead);   // [b,h,d,s]  (transposed for attention)
    __bf16* attnb = take(nX);      // [b,s,H]

    // 1) bf16 conversion; weights transposed once so GEMM staging is scatter-free.
    cvt_f32_bf16<<<4096, 256, 0, stream>>>(x, xb, (int)nX);
    cvt_transpose_bf16<<<dim3(N_QKV / 64, HID / 64), 256, 0, stream>>>(
        W_qkv, wqkvt, HID, N_QKV);
    cvt_transpose_bf16<<<dim3(HID / 64, HID / 64), 256, 0, stream>>>(
        W_o, wot, HID, HID);

    // 2) fused QKV projection -> Q/K heads + transposed V heads
    gemm_bf16_wmma<1><<<dim3(N_QKV / 128, M_ROWS / 128), 128, 0, stream>>>(
        xb, wqkvt, b_qkv, M_ROWS, N_QKV, HID, nullptr, qb, kb, vbt);

    // 3) causal flash attention
    attn_flash_wmma<<<dim3(S_LEN / 64, 2 * NH), 128, 0, stream>>>(qb, kb, vbt, attnb);

    // 4) output projection -> f32
    gemm_bf16_wmma<0><<<dim3(HID / 128, M_ROWS / 128), 128, 0, stream>>>(
        attnb, wot, b_o, M_ROWS, HID, HID, out, nullptr, nullptr, nullptr);
}
